// BI_Intereaction_37744172598002
// MI455X (gfx1250) — matrix operand order repaired and checked by reading, not compiled, not run on hardware
//
#include <hip/hip_runtime.h>

// ---------------------------------------------------------------------------
// FM bi-interaction for MI455X (gfx1250), wave32 + v_wmma_f32_16x16x32_f16.
//   out[n,k] = 0.5 * ((x@E)^2 - (x*x)@(E*E))[n,k] * mask[n]
// HBM-bound: single streaming read of x (800 MB, non-temporal), E staged as
// pre-packed f16 WMMA-B fragments in workspace (L2-resident).
// ---------------------------------------------------------------------------

typedef _Float16 v16h __attribute__((ext_vector_type(16)));
typedef float    v8f  __attribute__((ext_vector_type(8)));
typedef float    v4f  __attribute__((ext_vector_type(4)));

#define N_ROWS   20000
#define F_DIM    10000
#define K_DIM    32
#define CHUNKS   313            // ceil(10000 / 32)
#define TILES    (N_ROWS / 16)  // 1250 row tiles, exact

// workspace layout (bytes)
#define MASK_OFF   0                                  // float[20000] = 80000 B
#define FRAG_DWORDS (CHUNKS * 2 * 32 * 8)             // 160256 dwords / table
#define FRAGL_OFF  81920                              // 32B-aligned
#define FRAGQ_OFF  (FRAGL_OFF + FRAG_DWORDS * 4)      // total ws ~1.37 MB

// ---------------------------------------------------------------------------
// mask helpers
// ---------------------------------------------------------------------------
__global__ void zero_mask_kernel(float* __restrict__ mask) {
    int i = blockIdx.x * blockDim.x + threadIdx.x;
    if (i < N_ROWS) mask[i] = 0.0f;
}

__global__ void scatter_mask_kernel(const int* __restrict__ idx,
                                    float* __restrict__ mask, int n) {
    int i = blockIdx.x * blockDim.x + threadIdx.x;
    if (i < n) {
        int r = idx[i];
        if (r >= 0 && r < N_ROWS) mask[r] = 1.0f;   // duplicate writes benign
    }
}

// ---------------------------------------------------------------------------
// Pre-pack E and E*E (f32 -> f16) into WMMA B-fragment register order.
// B is 32x16 (KxN) f16: lane = {n = lane&15, K-half = lane>>4}; VGPR v holds
// K = Khalf*8 + (v>>2)*16 + 2(v&3) (low 16 bits) and +1 (high 16 bits). One dword per thread.
// Fragment dword index: ((ch*2 + kh)*32 + lane)*8 + v
// ---------------------------------------------------------------------------
__global__ void build_frags_kernel(const float* __restrict__ emb,
                                   unsigned int* __restrict__ fragL,
                                   unsigned int* __restrict__ fragQ) {
    int t = blockIdx.x * blockDim.x + threadIdx.x;
    if (t >= FRAG_DWORDS) return;
    int v    = t & 7;
    int lane = (t >> 3) & 31;
    int kh   = (t >> 8) & 1;
    int ch   = t >> 9;

    int n   = kh * 16 + (lane & 15);          // output column 0..31
    int kk  = (lane >> 4) * 8 + (v >> 2) * 16 + 2 * (v & 3);  // K position within 32-chunk
    int f0  = ch * 32 + kk;

    float e0 = (f0     < F_DIM) ? emb[(long)f0       * K_DIM + n] : 0.0f;
    float e1 = (f0 + 1 < F_DIM) ? emb[(long)(f0 + 1) * K_DIM + n] : 0.0f;

    unsigned short hl0 = __builtin_bit_cast(unsigned short, (_Float16)e0);
    unsigned short hl1 = __builtin_bit_cast(unsigned short, (_Float16)e1);
    unsigned short hq0 = __builtin_bit_cast(unsigned short, (_Float16)(e0 * e0));
    unsigned short hq1 = __builtin_bit_cast(unsigned short, (_Float16)(e1 * e1));

    fragL[t] = (unsigned int)hl0 | ((unsigned int)hl1 << 16);
    fragQ[t] = (unsigned int)hq0 | ((unsigned int)hq1 << 16);
}

// ---------------------------------------------------------------------------
// Main kernel: 8 waves/block, one 16-row tile per wave, K-reduction over F in
// 32-wide chunks. 4 WMMAs per chunk (left/right x two 16-col halves).
// ---------------------------------------------------------------------------
__global__ __launch_bounds__(256) void fm_wmma_kernel(
    const float* __restrict__ x,
    const float* __restrict__ mask,
    const unsigned int* __restrict__ fragL,
    const unsigned int* __restrict__ fragQ,
    float* __restrict__ out) {

    const int lane = threadIdx.x & 31;
    const int wave = threadIdx.x >> 5;
    const int tile = blockIdx.x * 8 + wave;
    if (tile >= TILES) return;                 // wave-uniform: EXEC stays full

    const int r0 = tile * 16;
    const int m  = lane & 15;                  // A-matrix row within tile
    const int kb = (lane >> 4) * 8;            // A-fragment K sub-base
    const float* xrow = x + (long)(r0 + m) * F_DIM;

    v8f accl0 = {}, accl1 = {}, accq0 = {}, accq1 = {};

    #pragma unroll 2
    for (int ch = 0; ch < CHUNKS; ++ch) {
        const int f0 = ch * 32;

        // ---- A fragments: x and x^2 (square in f32, round once to f16) ----
        // lane holds K = {kb..kb+7} U {kb+16..kb+23}: two aligned 32B runs.
        v4f s0a = {}, s0b = {}, s1a = {}, s1b = {};
        const int fs0 = f0 + kb;
        const int fs1 = f0 + kb + 16;
        if (fs0 + 8 <= F_DIM) {
            const v4f* p = reinterpret_cast<const v4f*>(xrow + fs0);
            s0a = __builtin_nontemporal_load(p);
            s0b = __builtin_nontemporal_load(p + 1);
        }
        if (fs1 + 8 <= F_DIM) {
            const v4f* p = reinterpret_cast<const v4f*>(xrow + fs1);
            s1a = __builtin_nontemporal_load(p);
            s1b = __builtin_nontemporal_load(p + 1);
        }
        if (ch + 1 < CHUNKS)                   // global_prefetch_b8 next chunk
            __builtin_prefetch(xrow + f0 + 32 + kb, 0, 0);

        float seg[16];
        #pragma unroll
        for (int j = 0; j < 4; ++j) {
            seg[j]      = s0a[j];
            seg[4 + j]  = s0b[j];
            seg[8 + j]  = s1a[j];
            seg[12 + j] = s1b[j];
        }
        v16h a_l, a_q;
        #pragma unroll
        for (int j = 0; j < 16; ++j) {
            float v = seg[j];
            a_l[j] = (_Float16)v;
            a_q[j] = (_Float16)(v * v);
        }

        // ---- B fragments: contiguous 32B per lane, L2-resident ----
        const long base = ((long)ch * 2) * 32 * 8;
        v16h bl0 = *reinterpret_cast<const v16h*>(fragL + base + (long)lane * 8);
        v16h bl1 = *reinterpret_cast<const v16h*>(fragL + base + 256 + (long)lane * 8);
        v16h bq0 = *reinterpret_cast<const v16h*>(fragQ + base + (long)lane * 8);
        v16h bq1 = *reinterpret_cast<const v16h*>(fragQ + base + 256 + (long)lane * 8);

        // ---- 4x v_wmma_f32_16x16x32_f16 ----
        accl0 = __builtin_amdgcn_wmma_f32_16x16x32_f16(false, a_l, false, bl0,
                                                       (short)0, accl0, false, false);
        accl1 = __builtin_amdgcn_wmma_f32_16x16x32_f16(false, a_l, false, bl1,
                                                       (short)0, accl1, false, false);
        accq0 = __builtin_amdgcn_wmma_f32_16x16x32_f16(false, a_q, false, bq0,
                                                       (short)0, accq0, false, false);
        accq1 = __builtin_amdgcn_wmma_f32_16x16x32_f16(false, a_q, false, bq1,
                                                       (short)0, accq1, false, false);
    }

    // ---- epilogue: C/D layout -> 0.5*(l*l - q) * mask, non-temporal store ----
    const int n     = lane & 15;
    const int rbase = r0 + (lane >> 4) * 8;
    #pragma unroll
    for (int i = 0; i < 8; ++i) {
        const int row = rbase + i;
        const float mk = mask[row];
        float o0 = 0.5f * (accl0[i] * accl0[i] - accq0[i]) * mk;
        float o1 = 0.5f * (accl1[i] * accl1[i] - accq1[i]) * mk;
        __builtin_nontemporal_store(o0, out + (long)row * K_DIM + n);
        __builtin_nontemporal_store(o1, out + (long)row * K_DIM + 16 + n);
    }
}

// ---------------------------------------------------------------------------
extern "C" void kernel_launch(void* const* d_in, const int* in_sizes, int n_in,
                              void* d_out, int out_size, void* d_ws, size_t ws_size,
                              hipStream_t stream) {
    const float* x   = (const float*)d_in[0];   // [20000, 10000] f32
    const float* emb = (const float*)d_in[1];   // [10000, 32]    f32
    const int*   idx = (const int*)d_in[2];     // [16000]        int
    const int n_idx  = in_sizes[2];

    char* ws = (char*)d_ws;
    float*        mask  = (float*)(ws + MASK_OFF);
    unsigned int* fragL = (unsigned int*)(ws + FRAGL_OFF);
    unsigned int* fragQ = (unsigned int*)(ws + FRAGQ_OFF);

    zero_mask_kernel<<<(N_ROWS + 255) / 256, 256, 0, stream>>>(mask);
    scatter_mask_kernel<<<(n_idx + 255) / 256, 256, 0, stream>>>(idx, mask, n_idx);
    build_frags_kernel<<<(FRAG_DWORDS + 255) / 256, 256, 0, stream>>>(emb, fragL, fragQ);

    const int blocks = (TILES + 7) / 8;         // 157 blocks x 8 waves
    fm_wmma_kernel<<<blocks, 256, 0, stream>>>(x, mask, fragL, fragQ, (float*)d_out);
}
